// SpatialGatherModule_18004502904988
// MI455X (gfx1250) — matrix operand order repaired and checked by reading, not compiled, not run on hardware
//
#include <hip/hip_runtime.h>
#include <hip/hip_bf16.h>
#include <stdint.h>

typedef __attribute__((ext_vector_type(16))) _Float16 v16h;
typedef __attribute__((ext_vector_type(8)))  float    v8f;

namespace sg {
constexpr int Bn   = 16;
constexpr int Cn   = 512;
constexpr int Kn   = 150;     // real keys
constexpr int HW   = 16384;   // 128*128
constexpr int BK   = Bn * Kn; // 2400 softmax rows
constexpr int KT   = 10;      // N tiles of 16 -> 160 padded keys
constexpr int MT   = 128;     // C rows per block
constexpr int SSPLIT = 8;     // spatial split (atomic accumulate)
constexpr int SCHUNK = HW / SSPLIT;   // 2048
constexpr int SSTEP  = 32;            // WMMA K per step
constexpr int NITER  = SCHUNK / SSTEP; // 64
constexpr int FSTR = 40;      // LDS row stride in f16 (16B-aligned rows)
constexpr int PSTR = 40;
}

union H16 { _Float16 h[16]; uint4 q[2]; };
union Frag { uint4 q[2]; v16h v; };

// ---------------- Kernel 1: per-row softmax stats (online max / sum-exp) ----
__global__ void __launch_bounds__(256) sg_rowstats(const float* __restrict__ probs,
                                                   float* __restrict__ stats) {
  using namespace sg;
  const int row = blockIdx.x;                       // 0 .. BK-1
  const float4* p4 = (const float4*)(probs + (size_t)row * HW);
  float m = -3.0e38f, s = 0.0f;
  for (int i = threadIdx.x; i < HW / 4; i += 256) {
    float4 v = p4[i];
    float xm = fmaxf(fmaxf(v.x, v.y), fmaxf(v.z, v.w));
    float nm = fmaxf(m, xm);
    s = s * __expf(m - nm)
      + __expf(v.x - nm) + __expf(v.y - nm) + __expf(v.z - nm) + __expf(v.w - nm);
    m = nm;
  }
  __shared__ float sm[256];
  __shared__ float ss[256];
  sm[threadIdx.x] = m; ss[threadIdx.x] = s;
  __syncthreads();
  for (int off = 128; off > 0; off >>= 1) {
    if (threadIdx.x < off) {
      float m2 = sm[threadIdx.x + off], s2 = ss[threadIdx.x + off];
      float m1 = sm[threadIdx.x],       s1 = ss[threadIdx.x];
      float M  = fmaxf(m1, m2);
      ss[threadIdx.x] = s1 * __expf(m1 - M) + s2 * __expf(m2 - M);
      sm[threadIdx.x] = M;
    }
    __syncthreads();
  }
  if (threadIdx.x == 0) { stats[row] = sm[0]; stats[BK + row] = ss[0]; }
}

// ---------------- Kernel 2: zero output ------------------------------------
__global__ void __launch_bounds__(256) sg_zero(float* __restrict__ out, int n) {
  int i = blockIdx.x * 256 + threadIdx.x;
  if (i < n) out[i] = 0.0f;
}

// ---------------- Kernel 3: softmax-normalize + f16 WMMA GEMM ---------------
// out[b,c,k] += sum_s p16[k,s] * f16[c,s]  (f32 accumulate, atomic over S-split)
// Double-buffered LDS + register-staged global loads + 2-deep B-fragment pipe.
__global__ void __launch_bounds__(256) sg_wmma_gemm(const float* __restrict__ features,
                                                    const float* __restrict__ probs,
                                                    const float* __restrict__ stats,
                                                    float* __restrict__ out) {
  using namespace sg;
  __shared__ __align__(16) _Float16 ldsF[2][MT * FSTR];        // 2 x 128 x 32 (padded)
  __shared__ __align__(16) _Float16 ldsP[2][KT * 16 * PSTR];   // 2 x 160 x 32 (padded)

  const int t      = threadIdx.x;
  const int b      = blockIdx.z;
  const int cBase  = blockIdx.y * MT;
  const int sBase0 = blockIdx.x * SCHUNK;

  // ---- fill-stage assignments (fixed per thread across the S loop) ----
  const int frow = t >> 1, fhalf = t & 1;                   // features: 128 rows x 2 halves
  const float4* fp = (const float4*)(features +
      ((size_t)(b * Cn + cBase + frow) * HW + sBase0 + fhalf * 16));

  const int   prow0 = t >> 1, phalf = t & 1;                // probs rows 0..127
  const float4* pp0 = (const float4*)(probs +
      ((size_t)(b * Kn + prow0) * HW + sBase0 + phalf * 16));
  const float  pm0   = stats[b * Kn + prow0];
  const float  pinv0 = 16384.0f / stats[BK + b * Kn + prow0];   // 2^14 pre-scale

  const bool  task1    = (t < 64);                          // probs rows 128..159
  const int   prow1    = 128 + (t >> 1);
  const bool  row1Real = task1 && (prow1 < Kn);
  const float4* pp1 = row1Real
      ? (const float4*)(probs + ((size_t)(b * Kn + prow1) * HW + sBase0 + phalf * 16))
      : nullptr;
  const float  pm1   = row1Real ? stats[b * Kn + prow1] : 0.0f;
  const float  pinv1 = row1Real ? 16384.0f / stats[BK + b * Kn + prow1] : 0.0f;

  // zero the padded key rows (150..159) once in BOTH buffers; never rewritten
  if (task1 && !row1Real) {
    H16 z;
    #pragma unroll
    for (int j = 0; j < 16; ++j) z.h[j] = (_Float16)0.0f;
    #pragma unroll
    for (int bb = 0; bb < 2; ++bb) {
      *(uint4*)&ldsP[bb][prow1 * PSTR + phalf * 16 + 0] = z.q[0];
      *(uint4*)&ldsP[bb][prow1 * PSTR + phalf * 16 + 8] = z.q[1];
    }
  }

  // ---- compute-stage assignments (ISA fragment layouts, wave32) ----
  const int lane = t & 31, wave = t >> 5;
  const int r    = lane & 15;        // N column / M row within tile
  const int half = lane >> 4;        // lane group 0-15 vs 16-31
  // A 16x32 f16: lanes0-15 -> K 0-7 & 16-23 ; lanes16-31 -> K 8-15 & 24-31
  const int aOff0 = (wave * 16 + r) * FSTR + half * 8;
  const int aOff1 = aOff0 + 16;
  // B 32x16 f16: lanes0-15 -> K 0-7 & 16-23 ; lanes16-31 -> K 8-15 & 24-31 (N = lane%16)
  const int bOff  = r * PSTR + half * 8;

  v8f acc[KT];
  #pragma unroll
  for (int n = 0; n < KT; ++n)
    #pragma unroll
    for (int e = 0; e < 8; ++e) acc[n][e] = 0.0f;

  // register staging for the next tile
  float4 fs[4], ps0[4], ps1[4];

  auto loadRegs = [&]() {
    fs[0] = fp[0]; fs[1] = fp[1]; fs[2] = fp[2]; fs[3] = fp[3]; fp += 8;
    ps0[0] = pp0[0]; ps0[1] = pp0[1]; ps0[2] = pp0[2]; ps0[3] = pp0[3]; pp0 += 8;
    if (row1Real) {
      ps1[0] = pp1[0]; ps1[1] = pp1[1]; ps1[2] = pp1[2]; ps1[3] = pp1[3]; pp1 += 8;
    }
  };

  auto cvtStore = [&](int buf) {
    H16 hf;
    #pragma unroll
    for (int j = 0; j < 4; ++j) {
      hf.h[4*j+0] = (_Float16)fs[j].x; hf.h[4*j+1] = (_Float16)fs[j].y;
      hf.h[4*j+2] = (_Float16)fs[j].z; hf.h[4*j+3] = (_Float16)fs[j].w;
    }
    *(uint4*)&ldsF[buf][frow * FSTR + fhalf * 16 + 0] = hf.q[0];
    *(uint4*)&ldsF[buf][frow * FSTR + fhalf * 16 + 8] = hf.q[1];

    float x[16] = { ps0[0].x,ps0[0].y,ps0[0].z,ps0[0].w, ps0[1].x,ps0[1].y,ps0[1].z,ps0[1].w,
                    ps0[2].x,ps0[2].y,ps0[2].z,ps0[2].w, ps0[3].x,ps0[3].y,ps0[3].z,ps0[3].w };
    H16 hp;
    #pragma unroll
    for (int j = 0; j < 16; ++j) hp.h[j] = (_Float16)(__expf(x[j] - pm0) * pinv0);
    *(uint4*)&ldsP[buf][prow0 * PSTR + phalf * 16 + 0] = hp.q[0];
    *(uint4*)&ldsP[buf][prow0 * PSTR + phalf * 16 + 8] = hp.q[1];

    if (row1Real) {
      float y[16] = { ps1[0].x,ps1[0].y,ps1[0].z,ps1[0].w, ps1[1].x,ps1[1].y,ps1[1].z,ps1[1].w,
                      ps1[2].x,ps1[2].y,ps1[2].z,ps1[2].w, ps1[3].x,ps1[3].y,ps1[3].z,ps1[3].w };
      H16 hq;
      #pragma unroll
      for (int j = 0; j < 16; ++j) hq.h[j] = (_Float16)(__expf(y[j] - pm1) * pinv1);
      *(uint4*)&ldsP[buf][prow1 * PSTR + phalf * 16 + 0] = hq.q[0];
      *(uint4*)&ldsP[buf][prow1 * PSTR + phalf * 16 + 8] = hq.q[1];
    }
  };

  auto compute = [&](int buf) {
    const _Float16* Fb = ldsF[buf];
    const _Float16* Pb = ldsP[buf];
    Frag A;
    A.q[0] = *(const uint4*)&Fb[aOff0];
    A.q[1] = *(const uint4*)&Fb[aOff1];
    Frag Bb[2];                       // 2-deep B pipeline: prefetch n+1 under wmma n
    Bb[0].q[0] = *(const uint4*)&Pb[bOff + 0];
    Bb[0].q[1] = *(const uint4*)&Pb[bOff + 16];
    #pragma unroll
    for (int n = 0; n < KT; ++n) {
      if (n + 1 < KT) {
        Bb[(n + 1) & 1].q[0] = *(const uint4*)&Pb[bOff + (n + 1) * 16 * PSTR + 0];
        Bb[(n + 1) & 1].q[1] = *(const uint4*)&Pb[bOff + (n + 1) * 16 * PSTR + 16];
      }
      acc[n] = __builtin_amdgcn_wmma_f32_16x16x32_f16(
          /*neg_a=*/false, A.v, /*neg_b=*/false, Bb[n & 1].v,
          /*c_mod=*/(short)0, acc[n], /*reuse_a=*/false, /*reuse_b=*/false);
    }
  };

  // ---- pipelined main loop: one barrier per iteration ----
  loadRegs();
  cvtStore(0);
  __syncthreads();
  for (int it = 0; it < NITER - 1; ++it) {
    loadRegs();                 // global loads for tile it+1 (issued early)
    compute(it & 1);            // WMMAs on current LDS buffer
    cvtStore((it + 1) & 1);     // exp/cvt + LDS stores for next buffer
    __syncthreads();
  }
  compute((NITER - 1) & 1);

  // ---------- epilogue: undo 2^14 pre-scale, atomically accumulate ----------
  // D layout: VGPR v, lanes0-15 -> M=v, N=lane ; lanes16-31 -> M=v+8, N=lane-16
  const float sc = 1.0f / 16384.0f;
  const int cRow = cBase + wave * 16 + half * 8;
  #pragma unroll
  for (int n = 0; n < KT; ++n) {
    const int key = n * 16 + r;
    if (key < Kn) {
      #pragma unroll
      for (int v = 0; v < 8; ++v) {
        atomicAdd(&out[((size_t)(b * Cn + cRow + v)) * Kn + key], acc[n][v] * sc);
      }
    }
  }
}

extern "C" void kernel_launch(void* const* d_in, const int* in_sizes, int n_in,
                              void* d_out, int out_size, void* d_ws, size_t ws_size,
                              hipStream_t stream) {
  using namespace sg;
  const float* features = (const float*)d_in[0];
  const float* probs    = (const float*)d_in[1];
  float* out   = (float*)d_out;
  float* stats = (float*)d_ws;          // [BK] rowmax, [BK] rowsum

  sg_rowstats<<<BK, 256, 0, stream>>>(probs, stats);
  sg_zero<<<(out_size + 255) / 256, 256, 0, stream>>>(out, out_size);
  dim3 grid(SSPLIT, Cn / MT, Bn);       // 8 x 4 x 16 = 512 blocks, 8 waves each
  sg_wmma_gemm<<<grid, 256, 0, stream>>>(features, probs, stats, out);
}
